// LSTM_65609920414504
// MI455X (gfx1250) — matrix operand order repaired and checked by reading, not compiled, not run on hardware
//
#include <hip/hip_runtime.h>
#include <hip/hip_bf16.h>
#include <cstdint>
#include <cstddef>

#define H      9000
#define G4     36000   // 4*H gate rows
#define LDH    9216    // padded bf16 row stride for w_hh (multiple of 32*8)
#define LDI    320     // padded bf16 row stride for w_ih
#define EMBED  300
#define STEPS  20

typedef __attribute__((ext_vector_type(16))) __bf16 v16bf;
typedef __attribute__((ext_vector_type(8)))  float  v8f;

union BV   { uint4 u[2]; v16bf v; };     // 32B: 16 bf16 operand block
union AccV { v8f v; float f[8]; };       // 16x16 f32 D tile, per-lane slice

__device__ __forceinline__ unsigned short f2bf(float f) {
    unsigned int u = __float_as_uint(f);
    u += 0x7fffu + ((u >> 16) & 1u);       // round-to-nearest-even
    return (unsigned short)(u >> 16);
}
__device__ __forceinline__ float sigm(float v) { return 1.0f / (1.0f + expf(-v)); }

// ---- fp32 -> bf16 weight conversion (padded, zero-filled pads) -------------
__global__ void k_cvt_whh(const float* __restrict__ w, unsigned short* __restrict__ o) {
    const int k = blockIdx.x * 256 + threadIdx.x;   // [0, 9216)
    const int r = blockIdx.y;                       // [0, 36000)
    o[(size_t)r * LDH + k] = (k < H) ? f2bf(w[(size_t)r * H + k]) : (unsigned short)0;
}
__global__ void k_cvt_wih(const float* __restrict__ w, unsigned short* __restrict__ o) {
    const int k = threadIdx.x;                      // [0, 320)
    const int r = blockIdx.x;                       // [0, 36000)
    o[(size_t)r * LDI + k] = (k < EMBED) ? f2bf(w[(size_t)r * EMBED + k]) : (unsigned short)0;
}
__global__ void k_cvt_vec300(const float* __restrict__ v, unsigned short* __restrict__ o) {
    const int k = threadIdx.x;                      // [0, 320)
    o[k] = (k < EMBED) ? f2bf(v[k]) : (unsigned short)0;
}

// ---- WMMA bf16 gate GEMV: gates[36000] = Wih*vih + Whh*vhh + b_ih + b_hh ---
// Block = 256 threads (8 waves), one 16-row tile per block; waves K-split.
__global__ void k_gates_wmma(const __bf16* __restrict__ Whh,
                             const __bf16* __restrict__ Wih,
                             const __bf16* __restrict__ vhh,   // len LDH, pad zero
                             const __bf16* __restrict__ vih,   // len LDI, pad zero
                             const float* __restrict__ b_ih,
                             const float* __restrict__ b_hh,
                             float* __restrict__ gates,
                             int use_hh)
{
    __shared__ float red[128];
    const int tid  = threadIdx.x;
    const int lane = tid & 31;
    const int wave = tid >> 5;          // 0..7
    const int half = lane >> 4;         // 0/1: lane group (K offset in A/B layout)
    const int mrow = lane & 15;         // matrix row owned by this lane in A
    const size_t m0 = (size_t)blockIdx.x * 16;

    AccV acc;
#pragma unroll
    for (int g = 0; g < 8; ++g) acc.f[g] = 0.f;

    // w_ih contribution: K = 320 -> 10 chunks of 32, round-robin over waves
    {
        const __bf16* Ar = Wih + (m0 + mrow) * LDI;
        for (int kc = wave; kc < LDI / 32; kc += 8) {
            const int k0 = kc * 32;
            BV a, b;
            a.u[0] = *(const uint4*)(Ar + k0 + 8 * half);        // K 0-7 / 8-15
            a.u[1] = *(const uint4*)(Ar + k0 + 16 + 8 * half);   // K 16-23 / 24-31
            const uint4* pb = (const uint4*)(vih + k0 + 8 * half);
            b.u[0] = pb[0]; b.u[1] = pb[2];                      // x chunk, all N cols equal
            acc.v = __builtin_amdgcn_wmma_f32_16x16x32_bf16(false, a.v, false, b.v,
                                                            (short)0, acc.v, false, false);
        }
    }
    // w_hh contribution: K = 9216 -> 288 chunks of 32, 36 contiguous per wave
    if (use_hh) {
        const __bf16* Ar = Whh + (m0 + mrow) * LDH;
        const int kc0 = wave * (LDH / 32 / 8);
        for (int kc = kc0; kc < kc0 + LDH / 32 / 8; ++kc) {
            const int k0 = kc * 32;
            BV a, b;
            a.u[0] = *(const uint4*)(Ar + k0 + 8 * half);
            a.u[1] = *(const uint4*)(Ar + k0 + 16 + 8 * half);
            const uint4* pb = (const uint4*)(vhh + k0 + 8 * half);
            b.u[0] = pb[0]; b.u[1] = pb[2];
            acc.v = __builtin_amdgcn_wmma_f32_16x16x32_bf16(false, a.v, false, b.v,
                                                            (short)0, acc.v, false, false);
        }
    }
    // D columns are identical; lanes with mrow==0 hold rows half*8+g in VGPR g.
    if (mrow == 0) {
#pragma unroll
        for (int g = 0; g < 8; ++g) red[wave * 16 + half * 8 + g] = acc.f[g];
    }
    __syncthreads();
    if (tid < 16) {
        float s = 0.f;
#pragma unroll
        for (int w = 0; w < 8; ++w) s += red[w * 16 + tid];
        const size_t r = m0 + tid;
        gates[r] = s + b_ih[r] + b_hh[r];
    }
}

// ---- fp32 fallback gates (used only if d_ws is too small for bf16 copies) --
__global__ void k_gates_fb(const float* __restrict__ w_ih, const float* __restrict__ w_hh,
                           const float* __restrict__ vin, const float* __restrict__ h,
                           const float* __restrict__ b_ih, const float* __restrict__ b_hh,
                           float* __restrict__ gates, int use_hh)
{
    const int lane = threadIdx.x & 31;
    const int wave = threadIdx.x >> 5;
    const size_t r = (size_t)blockIdx.x * 8 + wave;
    float s = 0.f;
    const float* wr = w_ih + r * EMBED;
    for (int k = lane; k < EMBED; k += 32) s += wr[k] * vin[k];
    if (use_hh) {
        const float4* w4 = (const float4*)(w_hh + r * H);
        const float4* h4 = (const float4*)h;
        for (int q = lane; q < H / 4; q += 32) {
            float4 a = w4[q], b = h4[q];
            s += a.x * b.x + a.y * b.y + a.z * b.z + a.w * b.w;
        }
    }
    for (int m = 16; m > 0; m >>= 1) s += __shfl_xor(s, m, 32);
    if (lane == 0) gates[r] = s + b_ih[r] + b_hh[r];
}

// ---- LSTM cell elementwise ------------------------------------------------
__global__ void k_cell(const float* __restrict__ gates, float* __restrict__ c,
                       float* __restrict__ hraw, int prime) {
    const int j = blockIdx.x * 256 + threadIdx.x;
    if (j >= H) return;
    const float gi = gates[j], gf = gates[j + H], gg = gates[j + 2 * H], go = gates[j + 3 * H];
    const float c0 = prime ? 0.f : c[j];
    const float cn = sigm(gf) * c0 + sigm(gi) * tanhf(gg);
    c[j] = cn;
    hraw[j] = sigm(go) * tanhf(cn);
}

// ---- h = hraw (priming step: no softmax), also produce bf16 copy ----------
__global__ void k_seth(const float* __restrict__ hraw, float* __restrict__ h,
                       unsigned short* __restrict__ vbf) {
    const int i = blockIdx.x * 256 + threadIdx.x;   // [0, 9216)
    if (i < H)        { const float v = hraw[i]; h[i] = v; vbf[i] = f2bf(v); }
    else if (i < LDH) { vbf[i] = 0; }
}

// ---- softmax over H, write h (fp32 + bf16) and output column --------------
__global__ void k_softmax(const float* __restrict__ hraw, float* __restrict__ h,
                          unsigned short* __restrict__ vbf, float* __restrict__ dout,
                          int step) {
    __shared__ float sm[1024];
    const int t = threadIdx.x;
    float m = -3.4e38f;
    for (int i = t; i < H; i += 1024) m = fmaxf(m, hraw[i]);
    sm[t] = m; __syncthreads();
    for (int o = 512; o > 0; o >>= 1) { if (t < o) sm[t] = fmaxf(sm[t], sm[t + o]); __syncthreads(); }
    m = sm[0]; __syncthreads();
    float s = 0.f;
    for (int i = t; i < H; i += 1024) s += expf(hraw[i] - m);
    sm[t] = s; __syncthreads();
    for (int o = 512; o > 0; o >>= 1) { if (t < o) sm[t] += sm[t + o]; __syncthreads(); }
    const float inv = 1.0f / sm[0];
    for (int i = t; i < LDH; i += 1024) {
        if (i < H) {
            const float v = expf(hraw[i] - m) * inv;
            h[i] = v; vbf[i] = f2bf(v);
            dout[(size_t)i * STEPS + step] = v;   // out is [H, STEPS] row-major
        } else vbf[i] = 0;
    }
}

// ---- out = w_out @ h + b_out (300 rows, fp32, wave-per-row) ---------------
__global__ void k_outvec(const float* __restrict__ w_out, const float* __restrict__ h,
                         const float* __restrict__ b_out, float* __restrict__ out) {
    const int lane = threadIdx.x & 31;
    const int wave = threadIdx.x >> 5;
    const int r = blockIdx.x * 8 + wave;
    if (r >= EMBED) return;
    const float4* w4 = (const float4*)(w_out + (size_t)r * H);
    const float4* h4 = (const float4*)h;
    float s = 0.f;
    for (int q = lane; q < H / 4; q += 32) {
        float4 a = w4[q], b = h4[q];
        s += a.x * b.x + a.y * b.y + a.z * b.z + a.w * b.w;
    }
    for (int m = 16; m > 0; m >>= 1) s += __shfl_xor(s, m, 32);
    if (lane == 0) out[r] = s + b_out[r];
}

extern "C" void kernel_launch(void* const* d_in, const int* in_sizes, int n_in,
                              void* d_out, int out_size, void* d_ws, size_t ws_size,
                              hipStream_t stream) {
    (void)in_sizes; (void)n_in; (void)out_size;
    const float* x     = (const float*)d_in[0];
    const float* w_ih  = (const float*)d_in[1];
    const float* w_hh  = (const float*)d_in[2];
    const float* b_ih  = (const float*)d_in[3];
    const float* b_hh  = (const float*)d_in[4];
    const float* w_out = (const float*)d_in[5];
    const float* b_out = (const float*)d_in[6];
    float* out = (float*)d_out;

    const size_t SZ_WHH = (size_t)G4 * LDH * 2;   // 663,552,000 B
    const size_t SZ_WIH = (size_t)G4 * LDI * 2;   //  23,040,000 B
    const size_t SMALLS = 272768;
    const bool bf16path = ws_size >= SZ_WHH + SZ_WIH + SMALLS;

    char* ws   = (char*)d_ws;
    char* base = bf16path ? (ws + SZ_WHH + SZ_WIH) : ws;
    unsigned short* whh_bf = (unsigned short*)ws;
    unsigned short* wih_bf = (unsigned short*)(ws + SZ_WHH);
    unsigned short* vhh_bf = (unsigned short*)(base + 0);        // 9216 bf16
    unsigned short* vih_bf = (unsigned short*)(base + 18432);    //  320 bf16
    float* h     = (float*)(base + 19200);                       // 9000 f32
    float* c     = (float*)(base + 55296);                       // 9000 f32
    float* hraw  = (float*)(base + 91392);                       // 9000 f32
    float* ovec  = (float*)(base + 127488);                      //  300 f32
    float* gates = (float*)(base + 128768);                      // 36000 f32

    if (bf16path) {
        k_cvt_whh<<<dim3(LDH / 256, G4), 256, 0, stream>>>(w_hh, whh_bf);
        k_cvt_wih<<<G4, LDI, 0, stream>>>(w_ih, wih_bf);
        k_cvt_vec300<<<1, LDI, 0, stream>>>(x, vih_bf);
        k_gates_wmma<<<G4 / 16, 256, 0, stream>>>((const __bf16*)whh_bf, (const __bf16*)wih_bf,
            (const __bf16*)vhh_bf, (const __bf16*)vih_bf, b_ih, b_hh, gates, 0);
    } else {
        k_gates_fb<<<G4 / 8, 256, 0, stream>>>(w_ih, w_hh, x, h, b_ih, b_hh, gates, 0);
    }
    k_cell<<<(H + 255) / 256, 256, 0, stream>>>(gates, c, hraw, 1);
    k_seth<<<LDH / 256, 256, 0, stream>>>(hraw, h, vhh_bf);

    for (int step = 0; step < STEPS; ++step) {
        k_outvec<<<(EMBED + 7) / 8, 256, 0, stream>>>(w_out, h, b_out, ovec);
        if (bf16path) {
            k_cvt_vec300<<<1, LDI, 0, stream>>>(ovec, vih_bf);
            k_gates_wmma<<<G4 / 16, 256, 0, stream>>>((const __bf16*)whh_bf, (const __bf16*)wih_bf,
                (const __bf16*)vhh_bf, (const __bf16*)vih_bf, b_ih, b_hh, gates, 1);
        } else {
            k_gates_fb<<<G4 / 8, 256, 0, stream>>>(w_ih, w_hh, ovec, h, b_ih, b_hh, gates, 1);
        }
        k_cell<<<(H + 255) / 256, 256, 0, stream>>>(gates, c, hraw, 0);
        k_softmax<<<1, 1024, 0, stream>>>(hraw, h, vhh_bf, out, step);
    }
}
